// Insignificant_Attention_38998303047747
// MI455X (gfx1250) — matrix operand order repaired and checked by reading, not compiled, not run on hardware
//
#include <hip/hip_runtime.h>
#include <hip/hip_bf16.h>

// ---------------------------------------------------------------------------
// MI455X (gfx1250) fused attention, all-contiguous WMMA operand loads,
// 32-row tiles for 2x operand reuse:
//   cast f32->f16 (+weight transpose), 3 projections (WMMA, V stored
//   transposed), Ksum/argmin trick for c0, flash-style 2-pass softmax
//   attention (WMMA), row-c0 zeroing epilogue.
// All matrix math: v_wmma_f32_16x16x32_f16 (f32 accumulate).
// ---------------------------------------------------------------------------

typedef __attribute__((ext_vector_type(16))) _Float16 v16h;
typedef __attribute__((ext_vector_type(8)))  _Float16 v8h;
typedef __attribute__((ext_vector_type(8)))  float    v8f;

#define Bsz 8
#define Ssz 2048
#define Dsz 512
#define Osz 512
#define BS  (Bsz * Ssz)          // 16384
#define INV_SCALE 0.125f         // 1/sqrt(64)

__device__ __forceinline__ v8f wmma16(v16h a, v16h b, v8f c) {
  return __builtin_amdgcn_wmma_f32_16x16x32_f16(false, a, false, b,
                                                (short)0, c, false, false);
}

// A-frag 16x32 (MxK), src row-major, leading dim ld (halves).
// lane 0-15: row=lane, K = kb+{0..15}; lane 16-31: row=lane-16, K = kb+{16..31}
__device__ __forceinline__ v16h load_frag_a(const _Float16* tile, int ld,
                                            int kb, int lane) {
  int row = lane & 15;
  int h   = lane >> 4;
  const _Float16* p = tile + (size_t)row * ld + kb + h * 16;
  v8h lo = *(const v8h*)p;
  v8h hi = *(const v8h*)(p + 8);
  v16h a;
#pragma unroll
  for (int i = 0; i < 8; ++i) { a[i] = lo[i]; a[8 + i] = hi[i]; }
  return a;
}

// B-frag 32x16 (KxN) where B = M^T, M row-major [rows, ld]: B(k,n)=M[n0+n][kb+k].
// Fully contiguous per lane -> single 32B vector load.
__device__ __forceinline__ v16h load_frag_bT(const _Float16* m, int ld,
                                             int n0, int kb, int lane) {
  int n  = lane & 15;
  int kh = (lane >> 4) * 16;
  return *(const v16h*)(m + (size_t)(n0 + n) * ld + kb + kh);
}

// C/D frag 16x16 f32: VGPR g, lanes 0-15 -> row g, lanes 16-31 -> row 8+g.
__device__ __forceinline__ void store_frag_f16(_Float16* dst, int ld,
                                               v8f c, int lane) {
  int col = lane & 15;
  int rb  = (lane >> 4) * 8;
#pragma unroll
  for (int g = 0; g < 8; ++g)
    dst[(size_t)(rb + g) * ld + col] = (_Float16)c[g];
}

// Transposed f16 store of a 16x16 D-frag: dst[(col)*ldT + row]; per lane the
// 8 rows are contiguous -> one 16B store.
__device__ __forceinline__ void store_frag_f16_T(_Float16* dst, int ldT,
                                                 v8f c, int lane) {
  int col = lane & 15;
  int rb  = (lane >> 4) * 8;
  v8h v;
#pragma unroll
  for (int g = 0; g < 8; ++g) v[g] = (_Float16)c[g];
  *(v8h*)(dst + (size_t)col * ldT + rb) = v;
}

// ---------------------------------------------------------------------------
__global__ void cast_f32_to_f16_kernel(const float* __restrict__ src,
                                       _Float16* __restrict__ dst, int n) {
  int i = blockIdx.x * blockDim.x + threadIdx.x;
  if (i < n) dst[i] = (_Float16)src[i];
}

// Cast + transpose weights: src [3][D][O] f32 -> dst [3][O][D] f16.
__global__ void castT_weights_kernel(const float* __restrict__ src,
                                     _Float16* __restrict__ dst) {
  int i = blockIdx.x * blockDim.x + threadIdx.x;
  if (i >= 3 * Dsz * Osz) return;
  int mat = i / (Dsz * Osz);
  int r   = i % (Dsz * Osz);
  int d = r / Osz, o = r % Osz;
  dst[(size_t)mat * Osz * Dsz + (size_t)o * Dsz + d] = (_Float16)src[i];
}

// ---------------------------------------------------------------------------
// Projections. Block = one (mat, 32-row tile); A staged in LDS, 8 waves each
// compute a 32x64 slice (2x4 WMMA accs, weight B-frags reused across the two
// row halves). V (mat==2) is stored transposed.
__global__ void proj_kernel(const _Float16* __restrict__ xh,
                            const _Float16* __restrict__ whT,  // [3][O][D]
                            _Float16* __restrict__ Qh,
                            _Float16* __restrict__ Kh,
                            _Float16* __restrict__ VhT) {      // [B][O][S]
  __shared__ _Float16 As[32][Dsz];                             // 32 KB
  int lane = threadIdx.x & 31, wave = threadIdx.x >> 5;
  int mtile = blockIdx.x & 511;    // BS/32
  int mat   = blockIdx.x >> 9;     // 0..2
  const _Float16* A = xh + (size_t)mtile * 32 * Dsz;
  for (int i = threadIdx.x; i < 32 * Dsz / 8; i += blockDim.x)
    ((uint4*)&As[0][0])[i] = ((const uint4*)A)[i];
  __syncthreads();

  const _Float16* Wt = whT + (size_t)mat * Osz * Dsz;
  int o0 = wave * 64;
  v8f acc[2][4] = {};
#pragma unroll 2
  for (int k = 0; k < Dsz; k += 32) {
    v16h a0 = load_frag_a(&As[0][0],  Dsz, k, lane);
    v16h a1 = load_frag_a(&As[16][0], Dsz, k, lane);
#pragma unroll
    for (int n = 0; n < 4; ++n) {
      v16h b = load_frag_bT(Wt, Dsz, o0 + n * 16, k, lane);
      acc[0][n] = wmma16(a0, b, acc[0][n]);
      acc[1][n] = wmma16(a1, b, acc[1][n]);
    }
  }

  if (mat == 2) {
    int b  = mtile >> 6;              // S/32 = 64 tiles per batch
    int sl = (mtile & 63) * 32;
    _Float16* Db = VhT + (size_t)b * Osz * Ssz + sl;
#pragma unroll
    for (int r = 0; r < 2; ++r)
#pragma unroll
      for (int n = 0; n < 4; ++n)
        store_frag_f16_T(Db + (size_t)(o0 + n * 16) * Ssz + r * 16,
                         Ssz, acc[r][n], lane);
  } else {
    _Float16* D = (mat == 0 ? Qh : Kh) + (size_t)mtile * 32 * Osz + o0;
#pragma unroll
    for (int r = 0; r < 2; ++r)
#pragma unroll
      for (int n = 0; n < 4; ++n)
        store_frag_f16(D + (size_t)r * 16 * Osz + n * 16, Osz, acc[r][n], lane);
  }
}

// ---------------------------------------------------------------------------
// Ksum[o] = sum_t K[0,t,o]; also init packed argmin slot.
__global__ void ksum_kernel(const _Float16* __restrict__ Kh,
                            float* __restrict__ ksum,
                            unsigned long long* __restrict__ amin) {
  int o = blockIdx.x * blockDim.x + threadIdx.x;
  if (o == 0) *amin = ~0ull;
  if (o < Osz) {
    float s = 0.f;
    for (int t = 0; t < Ssz; ++t) s += (float)Kh[(size_t)t * Osz + o];
    ksum[o] = s;
  }
}

// row_sums[0,s] = Q[0,s,:]·Ksum; packed atomicMin -> (sortable key | index)
__global__ void rowargmin_kernel(const _Float16* __restrict__ Qh,
                                 const float* __restrict__ ksum,
                                 unsigned long long* __restrict__ amin) {
  int s = blockIdx.x * blockDim.x + threadIdx.x;
  if (s >= Ssz) return;
  float acc = 0.f;
  for (int o = 0; o < Osz; ++o)
    acc += (float)Qh[(size_t)s * Osz + o] * ksum[o];
  unsigned ub  = __float_as_uint(acc);
  unsigned key = (ub & 0x80000000u) ? ~ub : (ub | 0x80000000u);
  unsigned long long pk = ((unsigned long long)key << 32) | (unsigned)s;
  atomicMin(amin, pk);
}

// ---------------------------------------------------------------------------
// Pass A: per 32-row tile, online softmax stats (m, l) over all 2048 keys.
// Each K B-frag feeds two WMMAs (row halves).
__global__ void softmax_stats_kernel(const _Float16* __restrict__ Qh,
                                     const _Float16* __restrict__ Kh,
                                     float* __restrict__ row_m,
                                     float* __restrict__ row_l) {
  __shared__ _Float16 Qs[32][Dsz];                   // 32 KB
  __shared__ float sh_m[8][32], sh_l[8][32];
  int lane = threadIdx.x & 31, wave = threadIdx.x >> 5;
  int b = blockIdx.x >> 6, stile = blockIdx.x & 63;
  const _Float16* Qb = Qh + ((size_t)b * Ssz + stile * 32) * Osz;
  const _Float16* Kb = Kh + (size_t)b * Ssz * Osz;

  for (int i = threadIdx.x; i < 32 * Dsz / 8; i += blockDim.x)
    ((uint4*)&Qs[0][0])[i] = ((const uint4*)Qb)[i];
  __syncthreads();

  float m[2][8], l[2][8];
#pragma unroll
  for (int r = 0; r < 2; ++r)
#pragma unroll
    for (int g = 0; g < 8; ++g) { m[r][g] = -1e30f; l[r][g] = 0.f; }

  for (int tt = wave; tt < Ssz / 16; tt += 8) {
    v8f acc[2] = {};
#pragma unroll
    for (int k = 0; k < Osz; k += 32) {
      v16h bf = load_frag_bT(Kb, Osz, tt * 16, k, lane);
      v16h a0 = load_frag_a(&Qs[0][0],  Dsz, k, lane);
      v16h a1 = load_frag_a(&Qs[16][0], Dsz, k, lane);
      acc[0] = wmma16(a0, bf, acc[0]);
      acc[1] = wmma16(a1, bf, acc[1]);
    }
#pragma unroll
    for (int r = 0; r < 2; ++r) {
#pragma unroll
      for (int g = 0; g < 8; ++g) {
        float v = acc[r][g] * INV_SCALE;
        float tmax = v;
        tmax = fmaxf(tmax, __shfl_xor(tmax, 1, 32));
        tmax = fmaxf(tmax, __shfl_xor(tmax, 2, 32));
        tmax = fmaxf(tmax, __shfl_xor(tmax, 4, 32));
        tmax = fmaxf(tmax, __shfl_xor(tmax, 8, 32));
        float mnew = fmaxf(m[r][g], tmax);
        float e = __expf(v - mnew);
        e += __shfl_xor(e, 1, 32);
        e += __shfl_xor(e, 2, 32);
        e += __shfl_xor(e, 4, 32);
        e += __shfl_xor(e, 8, 32);
        l[r][g] = l[r][g] * __expf(m[r][g] - mnew) + e;
        m[r][g] = mnew;
      }
    }
  }
  if ((lane & 15) == 0) {
    int half = lane >> 4;
#pragma unroll
    for (int r = 0; r < 2; ++r)
#pragma unroll
      for (int g = 0; g < 8; ++g) {
        sh_m[wave][r * 16 + half * 8 + g] = m[r][g];
        sh_l[wave][r * 16 + half * 8 + g] = l[r][g];
      }
  }
  __syncthreads();
  if (threadIdx.x < 32) {
    int r = threadIdx.x;
    float ms = sh_m[0][r];
#pragma unroll
    for (int w = 1; w < 8; ++w) ms = fmaxf(ms, sh_m[w][r]);
    float ls = 0.f;
#pragma unroll
    for (int w = 0; w < 8; ++w) ls += sh_l[w][r] * __expf(sh_m[w][r] - ms);
    size_t idx = (size_t)b * Ssz + stile * 32 + r;
    row_m[idx] = ms;
    row_l[idx] = ls;
  }
}

// ---------------------------------------------------------------------------
// Pass B: O = softmax(QK^T/8) @ V with precomputed stats; zero row c0 on store.
// 32-row tiles; V consumed transposed so P@V B-frags are contiguous and each
// feeds two WMMAs.
__global__ void attn_out_kernel(const _Float16* __restrict__ Qh,
                                const _Float16* __restrict__ Kh,
                                const _Float16* __restrict__ VhT,
                                const float* __restrict__ row_m,
                                const float* __restrict__ row_l,
                                const unsigned long long* __restrict__ amin,
                                float* __restrict__ out) {
  __shared__ _Float16 Qs[32][Dsz];                   // 32 KB
  __shared__ _Float16 Ps[32][128];                   // 8 KB
  int lane = threadIdx.x & 31, wave = threadIdx.x >> 5;
  int b = blockIdx.x >> 6, stile = blockIdx.x & 63;
  int s0 = stile * 32;
  const _Float16* Qb  = Qh + ((size_t)b * Ssz + s0) * Osz;
  const _Float16* Kb  = Kh + (size_t)b * Ssz * Osz;
  const _Float16* VbT = VhT + (size_t)b * Osz * Ssz;

  for (int i = threadIdx.x; i < 32 * Dsz / 8; i += blockDim.x)
    ((uint4*)&Qs[0][0])[i] = ((const uint4*)Qb)[i];

  int rb = (lane >> 4) * 8;
  float mloc[2][8], invl[2][8];
#pragma unroll
  for (int r = 0; r < 2; ++r)
#pragma unroll
    for (int g = 0; g < 8; ++g) {
      size_t idx = (size_t)b * Ssz + s0 + r * 16 + rb + g;
      mloc[r][g] = row_m[idx];
      invl[r][g] = 1.f / row_l[idx];
    }
  v8f accO[2][4] = {};
  __syncthreads();

  for (int t0 = 0; t0 < Ssz; t0 += 128) {
    // step 1: this wave computes score cols [16*wave, 16*wave+16) for both
    // row halves; one K B-frag per k feeds two WMMAs.
    int tt = t0 + wave * 16;
    v8f acc[2] = {};
#pragma unroll
    for (int k = 0; k < Osz; k += 32) {
      v16h bf = load_frag_bT(Kb, Osz, tt, k, lane);
      v16h a0 = load_frag_a(&Qs[0][0],  Dsz, k, lane);
      v16h a1 = load_frag_a(&Qs[16][0], Dsz, k, lane);
      acc[0] = wmma16(a0, bf, acc[0]);
      acc[1] = wmma16(a1, bf, acc[1]);
    }
    __syncthreads();   // previous-iter P consumers done
#pragma unroll
    for (int r = 0; r < 2; ++r)
#pragma unroll
      for (int g = 0; g < 8; ++g) {
        float p = __expf(acc[r][g] * INV_SCALE - mloc[r][g]) * invl[r][g];
        Ps[r * 16 + rb + g][wave * 16 + (lane & 15)] = (_Float16)p;
      }
    __syncthreads();
    // step 2: this wave owns output cols [64*wave, 64*wave+64); each V B-frag
    // feeds two WMMAs (row halves).
#pragma unroll
    for (int kk = 0; kk < 4; ++kk) {
      v16h a0 = load_frag_a(&Ps[0][0],  128, kk * 32, lane);
      v16h a1 = load_frag_a(&Ps[16][0], 128, kk * 32, lane);
#pragma unroll
      for (int n = 0; n < 4; ++n) {
        v16h bf = load_frag_bT(VbT, Ssz, wave * 64 + n * 16, t0 + kk * 32, lane);
        accO[0][n] = wmma16(a0, bf, accO[0][n]);
        accO[1][n] = wmma16(a1, bf, accO[1][n]);
      }
    }
  }

  int c0 = (int)(unsigned)(*amin & 0xffffffffu);
  float* ob = out + ((size_t)b * Ssz + s0) * Osz + wave * 64;
#pragma unroll
  for (int r = 0; r < 2; ++r)
#pragma unroll
    for (int n = 0; n < 4; ++n)
#pragma unroll
      for (int g = 0; g < 8; ++g) {
        int row = r * 16 + rb + g;
        int col = n * 16 + (lane & 15);
        float v = (s0 + row == c0) ? 0.f : accO[r][n][g];
        ob[(size_t)row * Osz + col] = v;
      }
}

// ---------------------------------------------------------------------------
extern "C" void kernel_launch(void* const* d_in, const int* in_sizes, int n_in,
                              void* d_out, int out_size, void* d_ws, size_t ws_size,
                              hipStream_t stream) {
  const float* x    = (const float*)d_in[0];   // [8,2048,512]
  const float* kern = (const float*)d_in[1];   // [3,512,512]
  float* out = (float*)d_out;                  // [8,2048,512]

  char* ws = (char*)d_ws;
  size_t off = 0;
  auto alloc = [&](size_t bytes) -> void* {
    void* p = ws + off;
    off = (off + bytes + 255) & ~(size_t)255;
    return p;
  };
  const size_t XN = (size_t)BS * Dsz;          // 8,388,608
  const size_t KN = (size_t)3 * Dsz * Osz;     // 786,432
  _Float16* xh   = (_Float16*)alloc(XN * 2);
  _Float16* whT  = (_Float16*)alloc(KN * 2);
  _Float16* qkvh = (_Float16*)alloc((size_t)3 * BS * Osz * 2);
  float*    ksum = (float*)alloc(Osz * 4);
  unsigned long long* amin = (unsigned long long*)alloc(8);
  float*    rm   = (float*)alloc((size_t)BS * 4);
  float*    rl   = (float*)alloc((size_t)BS * 4);

  _Float16* Qh  = qkvh;
  _Float16* Kh  = qkvh + (size_t)BS * Osz;
  _Float16* VhT = qkvh + (size_t)2 * BS * Osz;  // [B][O][S]

  cast_f32_to_f16_kernel<<<(int)((XN + 255) / 256), 256, 0, stream>>>(x, xh, (int)XN);
  castT_weights_kernel<<<(int)((KN + 255) / 256), 256, 0, stream>>>(kern, whT);

  // 3 mats x 512 row-tiles (32 rows each); 8 waves/block, 64 cols/wave
  proj_kernel<<<3 * 512, 256, 0, stream>>>(xh, whT, Qh, Kh, VhT);

  ksum_kernel<<<2, 256, 0, stream>>>(Kh, ksum, amin);
  rowargmin_kernel<<<Ssz / 256, 256, 0, stream>>>(Qh, ksum, amin);

  softmax_stats_kernel<<<Bsz * (Ssz / 32), 256, 0, stream>>>(Qh, Kh, rm, rl);
  attn_out_kernel<<<Bsz * (Ssz / 32), 256, 0, stream>>>(Qh, Kh, VhT, rm, rl, amin, out);
}
